// VICRegLLoss_62148176773694
// MI455X (gfx1250) — matrix operand order repaired and checked by reading, not compiled, not run on hardware
//
#include <hip/hip_runtime.h>
#include <hip/hip_bf16.h>

typedef __attribute__((ext_vector_type(2)))  float  v2f;
typedef __attribute__((ext_vector_type(8)))  float  v8f;
typedef __attribute__((ext_vector_type(16))) __bf16 v16bf;
typedef __attribute__((ext_vector_type(8)))  __bf16 v8bf;

#define VB   64      // batch
#define VD   8192    // global feature dim
#define VNP  784     // 28*28 local points
#define VCF  256     // local channels
#define VTILE 32
#define VKCH  128
#define VLDPB 136    // padded LDS row stride (bf16 elems): 272B = 17*16, spreads banks

union BF16x16 { v16bf v; v8bf h[2]; };

__device__ __forceinline__ unsigned long long u64min(unsigned long long a, unsigned long long b) {
    return a < b ? a : b;
}

__device__ __forceinline__ unsigned short f2bf_rne(float x) {
    unsigned u = __float_as_uint(x);
    unsigned r = (u + 0x7FFFu + ((u >> 16) & 1u)) >> 16;
    return (unsigned short)r;
}
__device__ __forceinline__ float bf2f(unsigned short h) {
    return __uint_as_float(((unsigned)h) << 16);
}

// ---------------- global loss: per-feature column stats ----------------
__global__ void vcl_stats_kernel(const float* __restrict__ za, const float* __restrict__ zb,
                                 float* __restrict__ mean_a, float* __restrict__ mean_b,
                                 float* __restrict__ S) {
    int j = blockIdx.x * blockDim.x + threadIdx.x;
    if (j >= VD) return;
    float sa = 0.f, qa = 0.f, sb = 0.f, qb = 0.f, sd = 0.f;
    for (int k = 0; k < VB; ++k) {
        float va = za[k * VD + j], vb = zb[k * VD + j];
        sa += va; qa += va * va;
        sb += vb; qb += vb * vb;
        float d = va - vb; sd += d * d;
    }
    float ma = sa * (1.f / VB), mb = sb * (1.f / VB);
    mean_a[j] = ma; mean_b[j] = mb;
    float ca = qa - (float)VB * ma * ma;   // centered SSQ = diag of Xc^T Xc
    float cb = qb - (float)VB * mb * mb;
    float stda = sqrtf(ca * (1.f / 63.f) + 1e-4f);
    float stdb = sqrtf(cb * (1.f / 63.f) + 1e-4f);
    atomicAdd(S + 0, sd);
    atomicAdd(S + 1, fmaxf(1.f - stda, 0.f));
    atomicAdd(S + 2, fmaxf(1.f - stdb, 0.f));
    atomicAdd(S + 3, ca * ca);
    atomicAdd(S + 4, cb * cb);
}

// ---------------- global loss: ||Xc Xc^T||_F^2 via f32 WMMA (64x64, K=8192) ----------------
__global__ __launch_bounds__(512) void vcl_gram_kernel(const float* __restrict__ za,
                                                       const float* __restrict__ zb,
                                                       const float* __restrict__ mean_a,
                                                       const float* __restrict__ mean_b,
                                                       float* __restrict__ S) {
    const float* X  = (blockIdx.x == 0) ? za : zb;
    const float* mu = (blockIdx.x == 0) ? mean_a : mean_b;
    int t = threadIdx.x, lane = t & 31, wave = t >> 5;
    int m0 = (wave >> 2) * 16, n0 = (wave & 3) * 16;   // 4x4 tile grid over 64x64
    int row = lane & 15;
    int ko  = (lane >> 4) * 2;                          // f32 A/B frag K layout
    const float* pa = X + (size_t)(m0 + row) * VD + ko;
    const float* pb = X + (size_t)(n0 + row) * VD + ko;
    const float* pm = mu + ko;
    v8f c = {};
#pragma unroll 4
    for (int k = 0; k < VD; k += 4) {
        float u0 = pm[k], u1 = pm[k + 1];
        v2f a = { pa[k] - u0, pa[k + 1] - u1 };
        v2f b = { pb[k] - u0, pb[k + 1] - u1 };
        c = __builtin_amdgcn_wmma_f32_16x16x4_f32(false, a, false, b, (short)0, c, false, false);
    }
    float fro = 0.f;
#pragma unroll
    for (int r = 0; r < 8; ++r) fro += c[r] * c[r];
    for (int m = 16; m >= 1; m >>= 1) fro += __shfl_xor(fro, m, 32);
    if (lane == 0) atomicAdd(S + 5 + blockIdx.x, fro);
}

// ---------------- local: per-point squared row norms ----------------
__global__ void vcl_rownorm_kernel(const float* __restrict__ za, const float* __restrict__ zb,
                                   float* __restrict__ rna, float* __restrict__ rnb) {
    int i = blockIdx.x * blockDim.x + threadIdx.x;
    if (i >= VB * VNP) return;
    const float4* a4 = (const float4*)(za + (size_t)i * VCF);
    const float4* b4 = (const float4*)(zb + (size_t)i * VCF);
    float sa = 0.f, sb = 0.f;
    for (int c = 0; c < VCF / 4; ++c) {
        float4 u = a4[c], v = b4[c];
        sa += u.x * u.x + u.y * u.y + u.z * u.z + u.w * u.w;
        sb += v.x * v.x + v.y * v.y + v.z * v.z + v.w * v.w;
    }
    rna[i] = sa; rnb[i] = sb;
}

// ---------------- local: split-bf16 WMMA pairwise d2 tiles + packed u64 NN atomic-min ----------------
// S = A.B^T emulated in ~f32 precision: Ahi*Bhi + Ahi*Blo + Alo*Bhi (lo*lo dropped),
// using V_WMMA_F32_16X16X32_BF16 (K=32): 3 instrs per 32-K window vs 8 for f32 WMMA.
__global__ __launch_bounds__(128) void vcl_pairdist_nn_kernel(
    const float* __restrict__ za, const float* __restrict__ zb,
    const float* __restrict__ rna, const float* __restrict__ rnb,
    unsigned long long* __restrict__ nn_ab, unsigned long long* __restrict__ nn_ba) {
    __shared__ unsigned short Ahi[VTILE * VLDPB];
    __shared__ unsigned short Alo[VTILE * VLDPB];
    __shared__ unsigned short Bhi[VTILE * VLDPB];
    __shared__ unsigned short Blo[VTILE * VLDPB];
    int b  = blockIdx.z;
    int m0 = blockIdx.y * VTILE;
    int n0 = blockIdx.x * VTILE;
    int t = threadIdx.x, lane = t & 31, wave = t >> 5;
    int sm = (wave >> 1) * 16, sn = (wave & 1) * 16;   // 2x2 subtiles per 32x32 tile
    const float* Ab = za + (size_t)b * VNP * VCF;
    const float* Bb = zb + (size_t)b * VNP * VCF;
    v8f c = {};
    int lh = lane >> 4;
    int ar = sm + (lane & 15);                          // A frag row (M)
    int br = sn + (lane & 15);                          // B frag col (N) -> row of zb
    for (int kk = 0; kk < VCF; kk += VKCH) {
        __syncthreads();
#pragma unroll
        for (int i = 0; i < 8; ++i) {                  // 32x128 chunk, float4 per thread
            int e = t + i * 128;
            int row = e >> 5;
            int c4  = (e & 31) << 2;
            int gm = m0 + row, gn = n0 + row;
            float4 av = (gm < VNP) ? *(const float4*)(Ab + (size_t)gm * VCF + kk + c4)
                                   : make_float4(0.f, 0.f, 0.f, 0.f);
            float4 bv = (gn < VNP) ? *(const float4*)(Bb + (size_t)gn * VCF + kk + c4)
                                   : make_float4(0.f, 0.f, 0.f, 0.f);
            unsigned short h0 = f2bf_rne(av.x), h1 = f2bf_rne(av.y),
                           h2 = f2bf_rne(av.z), h3 = f2bf_rne(av.w);
            unsigned short l0 = f2bf_rne(av.x - bf2f(h0)), l1 = f2bf_rne(av.y - bf2f(h1)),
                           l2 = f2bf_rne(av.z - bf2f(h2)), l3 = f2bf_rne(av.w - bf2f(h3));
            *(uint2*)(Ahi + row * VLDPB + c4) =
                make_uint2((unsigned)h0 | ((unsigned)h1 << 16), (unsigned)h2 | ((unsigned)h3 << 16));
            *(uint2*)(Alo + row * VLDPB + c4) =
                make_uint2((unsigned)l0 | ((unsigned)l1 << 16), (unsigned)l2 | ((unsigned)l3 << 16));
            h0 = f2bf_rne(bv.x); h1 = f2bf_rne(bv.y); h2 = f2bf_rne(bv.z); h3 = f2bf_rne(bv.w);
            l0 = f2bf_rne(bv.x - bf2f(h0)); l1 = f2bf_rne(bv.y - bf2f(h1));
            l2 = f2bf_rne(bv.z - bf2f(h2)); l3 = f2bf_rne(bv.w - bf2f(h3));
            *(uint2*)(Bhi + row * VLDPB + c4) =
                make_uint2((unsigned)h0 | ((unsigned)h1 << 16), (unsigned)h2 | ((unsigned)h3 << 16));
            *(uint2*)(Blo + row * VLDPB + c4) =
                make_uint2((unsigned)l0 | ((unsigned)l1 << 16), (unsigned)l2 | ((unsigned)l3 << 16));
        }
        __syncthreads();
        const __bf16* pAh = (const __bf16*)Ahi + ar * VLDPB;
        const __bf16* pAl = (const __bf16*)Alo + ar * VLDPB;
        const __bf16* pBh = (const __bf16*)Bhi + br * VLDPB;
        const __bf16* pBl = (const __bf16*)Blo + br * VLDPB;
#pragma unroll
        for (int kw = 0; kw < VKCH; kw += 32) {
            // A frag (16x32, 16-bit): lane holds K runs [8*lh, +8) and [16+8*lh, +8)
            BF16x16 ah, al, bh, bl;
            ah.h[0] = *(const v8bf*)(pAh + kw + 8 * lh);
            ah.h[1] = *(const v8bf*)(pAh + kw + 16 + 8 * lh);
            al.h[0] = *(const v8bf*)(pAl + kw + 8 * lh);
            al.h[1] = *(const v8bf*)(pAl + kw + 16 + 8 * lh);
            // B frag (32x16, 16-bit): lane holds column n, K runs [8*lh, +8) and [16+8*lh, +8)
            bh.h[0] = *(const v8bf*)(pBh + kw + 8 * lh);
            bh.h[1] = *(const v8bf*)(pBh + kw + 16 + 8 * lh);
            bl.h[0] = *(const v8bf*)(pBl + kw + 8 * lh);
            bl.h[1] = *(const v8bf*)(pBl + kw + 16 + 8 * lh);
            c = __builtin_amdgcn_wmma_f32_16x16x32_bf16(false, ah.v, false, bh.v, (short)0, c, false, false);
            c = __builtin_amdgcn_wmma_f32_16x16x32_bf16(false, ah.v, false, bl.v, (short)0, c, false, false);
            c = __builtin_amdgcn_wmma_f32_16x16x32_bf16(false, al.v, false, bh.v, (short)0, c, false, false);
        }
    }
    // epilogue: d2 = |a|^2 + |b|^2 - 2 a.b, clamp >= 0; pack (bits<<32 | idx), min-reduce
    int nl = lane & 15;
    int gn = n0 + sn + nl;
    float rb = (gn < VNP) ? rnb[b * VNP + gn] : 0.f;
    unsigned long long colmin = ~0ULL;
#pragma unroll
    for (int r = 0; r < 8; ++r) {
        int gm = m0 + sm + r + 8 * lh;
        unsigned long long prow = ~0ULL;
        if (gm < VNP && gn < VNP) {
            float d2 = fmaxf(rna[b * VNP + gm] + rb - 2.f * c[r], 0.f);
            unsigned long long bits = (unsigned long long)__float_as_uint(d2);
            prow = (bits << 32) | (unsigned)gn;
            colmin = u64min(colmin, (bits << 32) | (unsigned)gm);
        }
        for (int m = 1; m < 16; m <<= 1)
            prow = u64min(prow, __shfl_xor(prow, m, 32));
        if (nl == 0 && gm < VNP)
            atomicMin(nn_ab + (size_t)b * VNP + gm, prow);
    }
    colmin = u64min(colmin, __shfl_xor(colmin, 16, 32));
    if (lane < 16 && gn < VNP)
        atomicMin(nn_ba + (size_t)b * VNP + gn, colmin);
}

// ---------------- local: location-space NN (C=2, scalar) ----------------
__global__ void vcl_locnn_kernel(const float* __restrict__ la, const float* __restrict__ lb,
                                 unsigned long long* __restrict__ nn_ab,
                                 unsigned long long* __restrict__ nn_ba) {
    int i = blockIdx.x * blockDim.x + threadIdx.x;
    const int total = VB * VNP;
    if (i >= 2 * total) return;
    int dir = (i >= total) ? 1 : 0;
    int r = i - dir * total;
    int b = r / VNP, m = r % VNP;
    const float* inp = dir ? lb : la;
    const float* cnd = dir ? la : lb;
    float x = inp[((size_t)b * VNP + m) * 2 + 0];
    float y = inp[((size_t)b * VNP + m) * 2 + 1];
    const float* cb = cnd + (size_t)b * VNP * 2;
    unsigned long long best = ~0ULL;
    for (int n = 0; n < VNP; ++n) {
        float dx = x - cb[2 * n], dy = y - cb[2 * n + 1];
        float d2 = dx * dx + dy * dy;
        unsigned long long p = ((unsigned long long)__float_as_uint(d2) << 32) | (unsigned)n;
        best = u64min(best, p);
    }
    (dir ? nn_ba : nn_ab)[(size_t)b * VNP + m] = best;
}

// ---------------- local: rank-select top-K smallest, accumulate MSE sums ----------------
__global__ __launch_bounds__(256) void vcl_select_mse_kernel(
    const float* __restrict__ za, const float* __restrict__ zb,
    const unsigned long long* __restrict__ nnf_ab, const unsigned long long* __restrict__ nnf_ba,
    const unsigned long long* __restrict__ nnl_ab, const unsigned long long* __restrict__ nnl_ba,
    float* __restrict__ S) {
    __shared__ unsigned long long P[VNP];
    __shared__ float red[8];
    int b = blockIdx.x, dir = blockIdx.y;
    const unsigned long long* nn = (dir == 0) ? nnf_ab : (dir == 1) ? nnf_ba
                                 : (dir == 2) ? nnl_ab : nnl_ba;
    int K = (dir & 1) ? 4 : 20;
    const float* fin = (dir & 1) ? zb : za;   // input feature maps
    const float* fcn = (dir & 1) ? za : zb;   // candidate feature maps
    fin += (size_t)b * VNP * VCF;
    fcn += (size_t)b * VNP * VCF;
    nn  += (size_t)b * VNP;
    int t = threadIdx.x;
    for (int i = t; i < VNP; i += 256) P[i] = nn[i];
    __syncthreads();
    float lsum = 0.f;
    for (int i = t; i < VNP; i += 256) {
        unsigned long long pi = P[i];
        int rank = 0;
        for (int j = 0; j < VNP; ++j) rank += (P[j] < pi) ? 1 : 0;
        if (rank < K) {
            int idx = (int)(pi & 0xffffffffu);
            const float4* fa = (const float4*)(fin + (size_t)i * VCF);
            const float4* fb = (const float4*)(fcn + (size_t)idx * VCF);
            for (int c4 = 0; c4 < VCF / 4; ++c4) {
                float4 u = fa[c4], v = fb[c4];
                float dx = u.x - v.x, dy = u.y - v.y, dz = u.z - v.z, dw = u.w - v.w;
                lsum += dx * dx + dy * dy + dz * dz + dw * dw;
            }
        }
    }
    for (int m = 16; m >= 1; m >>= 1) lsum += __shfl_xor(lsum, m, 32);
    if ((t & 31) == 0) red[t >> 5] = lsum;
    __syncthreads();
    if (t == 0) {
        float s = 0.f;
        for (int w = 0; w < 8; ++w) s += red[w];
        atomicAdd(S + 7 + dir, s);
    }
}

// ---------------- finalize ----------------
__global__ void vcl_finalize_kernel(const float* __restrict__ S, float* __restrict__ out) {
    if (threadIdx.x == 0 && blockIdx.x == 0) {
        const float Bf = (float)VB, Df = (float)VD;
        float inv_g = S[0] / (Bf * Df);
        float var_g = 0.5f * (S[1] / Df + S[2] / Df);
        float denom = 63.f * 63.f * Df;            // (n-1)^2 * d
        float cov_g = (S[5] - S[3]) / denom + (S[6] - S[4]) / denom;
        float g = 25.f * inv_g + 25.f * var_g + 1.f * cov_g;
        float n20 = (float)VB * 20.f * (float)VCF;
        float n4  = (float)VB * 4.f * (float)VCF;
        float inv_l = 0.5f * (S[7] / n20 + S[8] / n4) + 0.5f * (S[9] / n20 + S[10] / n4);
        float l = 25.f * inv_l;
        out[0] = 0.25f * g + 0.75f * l;
    }
}

extern "C" void kernel_launch(void* const* d_in, const int* in_sizes, int n_in,
                              void* d_out, int out_size, void* d_ws, size_t ws_size,
                              hipStream_t stream) {
    const float* za  = (const float*)d_in[0];   // (64, 8192)
    const float* zb  = (const float*)d_in[1];
    const float* zal = (const float*)d_in[2];   // (64, 28, 28, 256)
    const float* zbl = (const float*)d_in[3];
    const float* la  = (const float*)d_in[4];   // (64, 28, 28, 2)
    const float* lb  = (const float*)d_in[5];

    char* ws = (char*)d_ws;                     // needs ~2.1 MB
    float* S      = (float*)ws;                                    // 16 f32 accumulators
    float* mean_a = (float*)(ws + 256);                            // 8192 f32
    float* mean_b = (float*)(ws + 256 + VD * 4);                   // 8192 f32
    size_t nn_off = 256 + 2 * (size_t)VD * 4;                      // 65792, 8B aligned
    unsigned long long* nnf_ab = (unsigned long long*)(ws + nn_off);
    unsigned long long* nnf_ba = nnf_ab + (size_t)VB * VNP;
    unsigned long long* nnl_ab = nnf_ba + (size_t)VB * VNP;
    unsigned long long* nnl_ba = nnl_ab + (size_t)VB * VNP;
    float* rna = (float*)(ws + nn_off + 4 * (size_t)VB * VNP * 8);
    float* rnb = rna + (size_t)VB * VNP;

    hipMemsetAsync(S, 0, 64, stream);
    hipMemsetAsync(nnf_ab, 0xFF, 4 * (size_t)VB * VNP * 8, stream);  // packed +inf

    vcl_stats_kernel<<<VD / 256, 256, 0, stream>>>(za, zb, mean_a, mean_b, S);
    vcl_gram_kernel<<<2, 512, 0, stream>>>(za, zb, mean_a, mean_b, S);
    vcl_rownorm_kernel<<<(VB * VNP + 255) / 256, 256, 0, stream>>>(zal, zbl, rna, rnb);
    dim3 gd((VNP + VTILE - 1) / VTILE, (VNP + VTILE - 1) / VTILE, VB);
    vcl_pairdist_nn_kernel<<<gd, 128, 0, stream>>>(zal, zbl, rna, rnb, nnf_ab, nnf_ba);
    vcl_locnn_kernel<<<(2 * VB * VNP + 255) / 256, 256, 0, stream>>>(la, lb, nnl_ab, nnl_ba);
    dim3 gs(VB, 4);
    vcl_select_mse_kernel<<<gs, 256, 0, stream>>>(zal, zbl, nnf_ab, nnf_ba, nnl_ab, nnl_ba, S);
    vcl_finalize_kernel<<<1, 64, 0, stream>>>(S, (float*)d_out);
}
